// CausalSelfAttention_82798379532432
// MI455X (gfx1250) — matrix operand order repaired and checked by reading, not compiled, not run on hardware
//
#include <hip/hip_runtime.h>

// ---------------------------------------------------------------------------
// CausalSelfAttention forward for MI455X (gfx1250), wave32 + WMMA f16 +
// async global->LDS double-buffered staging (ASYNCcnt path).
// B=4, T=2048, C=1024, H=16, HD=64.
// ---------------------------------------------------------------------------

typedef _Float16 half_t;
typedef __attribute__((ext_vector_type(16))) _Float16 v16h;
typedef __attribute__((ext_vector_type(8)))  _Float16 v8h;
typedef __attribute__((ext_vector_type(4)))  _Float16 v4h;
typedef __attribute__((ext_vector_type(8)))  float    v8f;
typedef __attribute__((ext_vector_type(4)))  int      v4i;

#define ATT_B 4
#define ATT_T 2048
#define ATT_C 1024
#define ATT_H 16
#define ATT_HD 64
#define ATT_PITCH 72   // 64 + 8 halves pad -> 144B rows (16B aligned chunks)
#define GM_TILE 128
#define GM_KSTEP 32
#define GM_PITCH 40    // 32 + 8 halves pad -> 80B rows (16B aligned chunks)

// ---- async global->LDS copy (CDNA5 ASYNCcnt path) -------------------------
#ifdef __has_builtin
#  if __has_builtin(__builtin_amdgcn_global_load_async_to_lds_b128)
#    define HAVE_ASYNC_BUILTIN 1
#  endif
#  if __has_builtin(__builtin_amdgcn_s_wait_asynccnt)
#    define HAVE_WAIT_BUILTIN 1
#  endif
#endif
#ifndef HAVE_ASYNC_BUILTIN
#  define HAVE_ASYNC_BUILTIN 0
#endif
#ifndef HAVE_WAIT_BUILTIN
#  define HAVE_WAIT_BUILTIN 0
#endif

typedef __attribute__((address_space(1))) v4i as1_v4i;   // global int4
typedef __attribute__((address_space(3))) v4i as3_v4i;   // LDS int4

// Move 16 bytes per lane: LDS[lptr] = MEM[gptr], tracked by ASYNCcnt.
__device__ __forceinline__ void ld_async_b128(void* lptr, const void* gptr) {
#if HAVE_ASYNC_BUILTIN
    __builtin_amdgcn_global_load_async_to_lds_b128(
        (as1_v4i*)(unsigned long long)gptr,
        (as3_v4i*)(unsigned)(unsigned long long)lptr,  // low 32b = LDS offset
        0, 0);
#else
    unsigned l = (unsigned)(unsigned long long)lptr;   // generic LDS ptr low 32b
    asm volatile("global_load_async_to_lds_b128 %0, %1, off"
                 :: "v"(l), "v"(gptr) : "memory");
#endif
}

__device__ __forceinline__ void wait_async_all() {
#if HAVE_WAIT_BUILTIN
    __builtin_amdgcn_s_wait_asynccnt(0);
#else
    asm volatile("s_wait_asynccnt 0x0" ::: "memory");
#endif
}

// ---- WMMA fragment helpers (ISA §7.12.2 layouts) --------------------------
__device__ __forceinline__ v16h ld_frag2(const half_t* p0, const half_t* p1) {
    v8h lo = *(const v8h*)p0;
    v8h hi = *(const v8h*)p1;
    return __builtin_shufflevector(lo, hi, 0,1,2,3,4,5,6,7,8,9,10,11,12,13,14,15);
}
// A 16x32 f16 fragment: halves[0..7]=K[8g..8g+8), halves[8..15]=K[16+8g..+8)
__device__ __forceinline__ v16h ld_fragA(const half_t* row_k0, int g) {
    return ld_frag2(row_k0 + 8*g, row_k0 + 16 + 8*g);
}
// B 32x16 f16 fragment from BT[n][k] storage: halves[0..7]=K[8g..8g+8), halves[8..15]=K[16+8g..+8)
__device__ __forceinline__ v16h ld_fragB(const half_t* bt_row_k0, int g) {
    return ld_frag2(bt_row_k0 + 8*g, bt_row_k0 + 16 + 8*g);
}
__device__ __forceinline__ v8f wmma_f16(v16h a, v16h b, v8f c) {
    return __builtin_amdgcn_wmma_f32_16x16x32_f16(false, a, false, b,
                                                  (short)0, c, false, false);
}

// ---------------------------------------------------------------------------
// f32 -> f16 conversion, 4 elems/thread
// ---------------------------------------------------------------------------
__global__ __launch_bounds__(256)
void cvt_f32_f16_v4(const float* __restrict__ in, half_t* __restrict__ out, int n4) {
    int i = blockIdx.x * 256 + threadIdx.x;
    if (i < n4) {
        float4 f = ((const float4*)in)[i];
        v4h h;
        h[0] = (half_t)f.x; h[1] = (half_t)f.y; h[2] = (half_t)f.z; h[3] = (half_t)f.w;
        *(v4h*)(out + (size_t)i * 4) = h;
    }
}

// ---------------------------------------------------------------------------
// WMMA GEMM: C[M,N] = A[M,K] @ W[N,K]^T + bias (torch Linear semantics).
// 256 threads (8 waves), block tile 128x128, K-step 32, async double-buffer.
// Wave w: rows (w>>1)*32 + {0,16}, cols (w&1)*64 + t*16 -> 8 wmma / K-step.
// OUT_MODE: 0 = f32 row-major, 1 = f16 row-major,
//           2 = f16 per-head transposed V layout [B,H,HD,T].
// ---------------------------------------------------------------------------
template<int OUT_MODE>
__global__ __launch_bounds__(256)
void gemm_wmma(const half_t* __restrict__ A, const half_t* __restrict__ W,
               const float* __restrict__ bias, void* __restrict__ Cout,
               int M, int N, int K) {
    __shared__ __align__(16) half_t As[2][GM_TILE][GM_PITCH];
    __shared__ __align__(16) half_t Bs[2][GM_TILE][GM_PITCH];

    const int tid  = threadIdx.x;
    const int wave = tid >> 5;
    const int lane = tid & 31;
    const int g    = lane >> 4;
    const int l16  = lane & 15;
    const int mw   = wave >> 1;      // 0..3
    const int nw   = wave & 1;       // 0..1

    const int m0 = blockIdx.x * GM_TILE;
    const int n0 = blockIdx.y * GM_TILE;

    auto stage = [&](int buf, int k0) {
        #pragma unroll
        for (int c = 0; c < 2; ++c) {
            int idx = tid + c * 256;        // 0..511
            int r   = idx >> 2;             // 0..127
            int cc  = (idx & 3) * 8;        // 0,8,16,24
            ld_async_b128(&As[buf][r][cc], A + (size_t)(m0 + r) * K + k0 + cc);
            ld_async_b128(&Bs[buf][r][cc], W + (size_t)(n0 + r) * K + k0 + cc);
        }
    };

    v8f acc[2][4] = {};
    stage(0, 0);
    const int nk = K / GM_KSTEP;
    for (int kb = 0; kb < nk; ++kb) {
        const int cur = kb & 1;
        wait_async_all();
        __syncthreads();
        if (kb + 1 < nk) stage(cur ^ 1, (kb + 1) * GM_KSTEP);

        v16h a0 = ld_fragA(&As[cur][mw * 32 + l16][0], g);
        v16h a1 = ld_fragA(&As[cur][mw * 32 + 16 + l16][0], g);
        #pragma unroll
        for (int t = 0; t < 4; ++t) {
            v16h b = ld_fragB(&Bs[cur][nw * 64 + t * 16 + l16][0], g);
            acc[0][t] = wmma_f16(a0, b, acc[0][t]);
            acc[1][t] = wmma_f16(a1, b, acc[1][t]);
        }
    }

    // Epilogue: register v holds row 8g+v, column = lane%16 (C/D layout).
    #pragma unroll
    for (int mi = 0; mi < 2; ++mi) {
        #pragma unroll
        for (int t = 0; t < 4; ++t) {
            int n = n0 + nw * 64 + t * 16 + l16;
            float bv = bias[n];
            #pragma unroll
            for (int v = 0; v < 8; ++v) {
                int m = m0 + mw * 32 + mi * 16 + 8 * g + v;
                float val = acc[mi][t][v] + bv;
                if (OUT_MODE == 0) {
                    ((float*)Cout)[(size_t)m * N + n] = val;
                } else if (OUT_MODE == 1) {
                    ((half_t*)Cout)[(size_t)m * N + n] = (half_t)val;
                } else {
                    // m = b*T + tok ; n = h*HD + d  ->  [B,H,HD,T]
                    size_t b_  = (size_t)(m >> 11);
                    size_t tok = (size_t)(m & (ATT_T - 1));
                    size_t h_  = (size_t)(n >> 6);
                    size_t d_  = (size_t)(n & (ATT_HD - 1));
                    ((half_t*)Cout)[((b_ * ATT_H + h_) * ATT_HD + d_) * ATT_T + tok] =
                        (half_t)val;
                }
            }
        }
    }
}

// ---------------------------------------------------------------------------
// Flash attention: block per (b, h, 64-row q tile). 128 threads, 4 waves,
// each wave owns 16 q rows. K and V^T tiles async double-buffered over the
// causal KV loop; S/P kept in registers/LDS (never spilled to HBM).
// ---------------------------------------------------------------------------
__global__ __launch_bounds__(128)
void attn_fwd(const half_t* __restrict__ Q, const half_t* __restrict__ Kb,
              const half_t* __restrict__ Vt, const int* __restrict__ padmask,
              half_t* __restrict__ Y) {
    __shared__ __align__(16) half_t Qs [64][ATT_PITCH];
    __shared__ __align__(16) half_t Ks [2][64][ATT_PITCH];
    __shared__ __align__(16) half_t Vts[2][64][ATT_PITCH];   // [d][k]
    __shared__ __align__(16) half_t Ps [4][16][ATT_PITCH];   // per-wave P

    const int bx   = blockIdx.x;
    const int iq   = bx & 31;                 // T/64 q tiles
    const int h    = (bx >> 5) & 15;
    const int b    = bx >> 9;
    const int tid  = threadIdx.x;
    const int wave = tid >> 5;
    const int lane = tid & 31;
    const int g    = lane >> 4;
    const int l16  = lane & 15;

    const size_t headOff  = ((size_t)b * ATT_T) * ATT_C + (size_t)h * ATT_HD;
    const size_t vHeadOff = ((size_t)(b * ATT_H + h)) * ATT_HD * ATT_T;
    const int q0 = iq * 64;

    auto stageKV = [&](int buf, int j) {
        const int k0g = j * 64;
        #pragma unroll
        for (int c = 0; c < 4; ++c) {
            int idx = tid + c * 128;          // 0..511
            int r   = idx >> 3;               // 0..63
            int cc  = (idx & 7) * 8;          // 0..56
            ld_async_b128(&Ks[buf][r][cc],
                          Kb + headOff + (size_t)(k0g + r) * ATT_C + cc);
            ld_async_b128(&Vts[buf][r][cc],
                          Vt + vHeadOff + (size_t)r * ATT_T + k0g + cc);
        }
    };

    // Q tile (async, covered by the first wait+barrier)
    #pragma unroll
    for (int c = 0; c < 4; ++c) {
        int idx = tid + c * 128;
        int r   = idx >> 3;
        int cc  = (idx & 7) * 8;
        ld_async_b128(&Qs[r][cc], Q + headOff + (size_t)(q0 + r) * ATT_C + cc);
    }
    stageKV(0, 0);

    float m[8], l[8];
    v8f o[4] = {};
    #pragma unroll
    for (int v = 0; v < 8; ++v) { m[v] = -3.0e38f; l[v] = 0.0f; }
    const float scale = 0.125f;   // 1/sqrt(HD)

    for (int j = 0; j <= iq; ++j) {
        const int cur = j & 1;
        const int k0g = j * 64;
        wait_async_all();
        __syncthreads();
        if (j < iq) stageKV(cur ^ 1, j + 1);

        // S = Q_wave(16xHD) @ K^T -> 4 tiles over 64 kv cols
        v8f s[4] = {};
        #pragma unroll
        for (int kk = 0; kk < 64; kk += 32) {
            v16h a = ld_fragA(&Qs[wave * 16 + l16][kk], g);
            #pragma unroll
            for (int t = 0; t < 4; ++t) {
                v16h bfr = ld_fragB(&Ks[cur][t * 16 + l16][kk], g);
                s[t] = wmma_f16(a, bfr, s[t]);
            }
        }

        // Scale + causal + padding mask
        const int qrowBase = q0 + wave * 16 + 8 * g;
        #pragma unroll
        for (int t = 0; t < 4; ++t) {
            const int kg = k0g + t * 16 + l16;
            const int pm = padmask[b * ATT_T + kg];
            #pragma unroll
            for (int v = 0; v < 8; ++v) {
                float val = s[t][v] * scale;
                if (kg > qrowBase + v || pm == 0) val = -3.0e38f;
                s[t][v] = val;
            }
        }

        // Online softmax (row stats replicated across the 16-lane half-wave)
        #pragma unroll
        for (int v = 0; v < 8; ++v) {
            float x = fmaxf(fmaxf(s[0][v], s[1][v]), fmaxf(s[2][v], s[3][v]));
            #pragma unroll
            for (int off = 1; off < 16; off <<= 1) x = fmaxf(x, __shfl_xor(x, off, 32));
            float mn = fmaxf(m[v], x);
            float alpha = __expf(m[v] - mn);
            l[v] *= alpha;
            o[0][v] *= alpha; o[1][v] *= alpha; o[2][v] *= alpha; o[3][v] *= alpha;
            m[v] = mn;
        }
        #pragma unroll
        for (int v = 0; v < 8; ++v) {
            float rs = 0.0f;
            #pragma unroll
            for (int t = 0; t < 4; ++t) {
                float p = __expf(s[t][v] - m[v]);
                rs += p;
                Ps[wave][8 * g + v][t * 16 + l16] = (half_t)p;
            }
            #pragma unroll
            for (int off = 1; off < 16; off <<= 1) rs += __shfl_xor(rs, off, 32);
            l[v] += rs;
        }

        // O += P(16x64) @ V(64xHD) with B-fragments from V^T tile
        #pragma unroll
        for (int kk = 0; kk < 64; kk += 32) {
            v16h a = ld_fragA(&Ps[wave][l16][kk], g);
            #pragma unroll
            for (int t = 0; t < 4; ++t) {
                v16h bfr = ld_fragB(&Vts[cur][t * 16 + l16][kk], g);
                o[t] = wmma_f16(a, bfr, o[t]);
            }
        }
    }

    // Epilogue: y = O / l, f16 into [B,T,H,HD]-interleaved buffer
    #pragma unroll
    for (int t = 0; t < 4; ++t) {
        int d = t * 16 + l16;
        #pragma unroll
        for (int v = 0; v < 8; ++v) {
            int qg = q0 + wave * 16 + 8 * g + v;
            Y[headOff + (size_t)qg * ATT_C + d] = (half_t)(o[t][v] / l[v]);
        }
    }
}

// ---------------------------------------------------------------------------
extern "C" void kernel_launch(void* const* d_in, const int* in_sizes, int n_in,
                              void* d_out, int out_size, void* d_ws, size_t ws_size,
                              hipStream_t stream) {
    (void)in_sizes; (void)n_in; (void)out_size; (void)ws_size;
    const float* x  = (const float*)d_in[0];
    const float* Wk = (const float*)d_in[1];
    const float* bk = (const float*)d_in[2];
    const float* Wq = (const float*)d_in[3];
    const float* bq = (const float*)d_in[4];
    const float* Wv = (const float*)d_in[5];
    const float* bv = (const float*)d_in[6];
    const float* Wp = (const float*)d_in[7];
    const float* bp = (const float*)d_in[8];
    const int*   pm = (const int*)d_in[9];
    float* out = (float*)d_out;

    const int M = ATT_B * ATT_T;   // 8192
    const int C = ATT_C;           // 1024

    char* ws = (char*)d_ws;
    half_t* xh  = (half_t*)ws;  ws += (size_t)M * C * sizeof(half_t);
    half_t* wkh = (half_t*)ws;  ws += (size_t)C * C * sizeof(half_t);
    half_t* wqh = (half_t*)ws;  ws += (size_t)C * C * sizeof(half_t);
    half_t* wvh = (half_t*)ws;  ws += (size_t)C * C * sizeof(half_t);
    half_t* wph = (half_t*)ws;  ws += (size_t)C * C * sizeof(half_t);
    half_t* qh  = (half_t*)ws;  ws += (size_t)M * C * sizeof(half_t);
    half_t* kh  = (half_t*)ws;  ws += (size_t)M * C * sizeof(half_t);
    half_t* vt  = (half_t*)ws;  ws += (size_t)M * C * sizeof(half_t); // [B,H,HD,T]
    half_t* yh  = (half_t*)ws;  ws += (size_t)M * C * sizeof(half_t);

    // f32 -> f16 converts
    {
        int n4 = M * C / 4;
        cvt_f32_f16_v4<<<(n4 + 255) / 256, 256, 0, stream>>>(x, xh, n4);
        int w4 = C * C / 4;
        dim3 gw((w4 + 255) / 256);
        cvt_f32_f16_v4<<<gw, 256, 0, stream>>>(Wk, wkh, w4);
        cvt_f32_f16_v4<<<gw, 256, 0, stream>>>(Wq, wqh, w4);
        cvt_f32_f16_v4<<<gw, 256, 0, stream>>>(Wv, wvh, w4);
        cvt_f32_f16_v4<<<gw, 256, 0, stream>>>(Wp, wph, w4);
    }

    // QKV projections (Q,K f16 row-major; V f16 transposed per head)
    dim3 gg(M / GM_TILE, C / GM_TILE);   // 64 x 8
    gemm_wmma<1><<<gg, 256, 0, stream>>>(xh, wqh, bq, qh, M, C, C);
    gemm_wmma<1><<<gg, 256, 0, stream>>>(xh, wkh, bk, kh, M, C, C);
    gemm_wmma<2><<<gg, 256, 0, stream>>>(xh, wvh, bv, vt, M, C, C);

    // Flash attention -> yh (f16, [B,T,H,HD] interleaved == [M,C])
    attn_fwd<<<ATT_B * ATT_H * (ATT_T / 64), 128, 0, stream>>>(qh, kh, vt, pm, yh);

    // Output projection (f32 out + bias) straight into d_out
    gemm_wmma<0><<<gg, 256, 0, stream>>>(yh, wph, bp, out, M, C, C);
}
